// NearestNeighborMatcher_11166914970513
// MI455X (gfx1250) — matrix operand order repaired and checked by reading, not compiled, not run on hardware
//
#include <hip/hip_runtime.h>

#define DEVINL __device__ __forceinline__

constexpr int B = 4, N = 4096, M = 4096, D = 256;
constexpr int MP = M / 128;   // row-partial chunks per row  (32)
constexpr int NP = N / 64;    // col-partial chunks per col  (64)

typedef __attribute__((ext_vector_type(16))) __bf16 v16bf;
typedef __attribute__((ext_vector_type(8)))  __bf16 v8bf;
typedef __attribute__((ext_vector_type(8)))  float  v8f;

// ---------------- f32 -> bf16 hi/lo split (RNE) ----------------
__global__ __launch_bounds__(256) void cvt_split_kernel(
    const float* __restrict__ in, unsigned short* __restrict__ hi,
    unsigned short* __restrict__ lo, int n) {
  int i = blockIdx.x * 256 + threadIdx.x;
  if (i >= n) return;
  float x = in[i];
  unsigned u = __float_as_uint(x);
  unsigned h = (u + 0x7FFFu + ((u >> 16) & 1u)) >> 16;
  hi[i] = (unsigned short)h;
  float r = x - __uint_as_float(h << 16);
  unsigned v = __float_as_uint(r);
  lo[i] = (unsigned short)((v + 0x7FFFu + ((v >> 16) & 1u)) >> 16);
}

// ---------------- online softmax merge ----------------
DEVINL void osm_merge(float& mx, float& sm, int& ai, float mx2, float sm2, int ai2) {
  float m = fmaxf(mx, mx2);
  float s = sm * __expf(mx - m) + sm2 * __expf(mx2 - m);
  if (mx2 > mx || (mx2 == mx && ai2 < ai)) ai = ai2;
  mx = m; sm = s;
}
DEVINL void shfl_merge(float& mx, float& sm, int& ai, int mask) {
  float m2 = __shfl_xor(mx, mask, 32);
  float s2 = __shfl_xor(sm, mask, 32);
  int   a2 = __shfl_xor(ai, mask, 32);
  osm_merge(mx, sm, ai, m2, s2, a2);
}

// ---------------- WMMA fragment loads ----------------
// A-matrix 16x32 bf16 (ISA 7.12.2): lanes 0-15 row M=lane hold K{0..7,16..23},
// lanes 16-31 same rows hold K{8..15,24..31}. Each half is 8 contiguous bf16.
DEVINL v16bf load_frag_a(const unsigned short* base, int row, int k0, int lane) {
  const __bf16* p = (const __bf16*)(base + (size_t)row * D + k0 + ((lane >> 4) << 3));
  v8bf l = *(const v8bf*)p;
  v8bf h = *(const v8bf*)(p + 16);
  return __builtin_shufflevector(l, h, 0,1,2,3,4,5,6,7,8,9,10,11,12,13,14,15);
}
// B-matrix 32x16 bf16: column N=lane&15; lanes 0-15 hold K{0..7,16..23}, lanes 16-31
// hold K{8..15,24..31} of one desc1 row (we feed desc1^T).
DEVINL v16bf load_frag_b(const unsigned short* base, int col, int k0, int lane) {
  const __bf16* p = (const __bf16*)(base + (size_t)col * D + k0 + ((lane >> 4) << 3));
  v8bf l = *(const v8bf*)p;
  v8bf h = *(const v8bf*)(p + 16);
  return __builtin_shufflevector(l, h, 0,1,2,3,4,5,6,7,8,9,10,11,12,13,14,15);
}

// ---------------- GEMM + fused row/col online-softmax partials ----------------
// Block 256 thr = 8 waves; wave tile 32x32 (2x2 WMMA tiles); block tile 64(N)x128(M).
// acc += a_hi*b_hi + a_hi*b_lo + a_lo*b_hi  (fp32-like precision on bf16 path)
__global__ __launch_bounds__(256) void gemm_sim_kernel(
    const unsigned short* __restrict__ a_hi, const unsigned short* __restrict__ a_lo,
    const unsigned short* __restrict__ b_hi, const unsigned short* __restrict__ b_lo,
    float* __restrict__ sim,
    float* __restrict__ rp_mx, float* __restrict__ rp_sm, int* __restrict__ rp_ai,
    float* __restrict__ cp_mx, float* __restrict__ cp_sm, int* __restrict__ cp_ai) {
  const int bz    = blockIdx.z;
  const int n_blk = blockIdx.y * 64;
  const int m_blk = blockIdx.x * 128;
  const int tid   = threadIdx.x;
  const int lane  = tid & 31, wave = tid >> 5;
  const int nw = wave & 1, mw = wave >> 1;     // wave coords: 2 along N, 4 along M
  const int wn = nw * 32, wm = mw * 32;

  const size_t aoff = (size_t)bz * N * D;
  const size_t boff = (size_t)bz * M * D;
  const unsigned short* Ah = a_hi + aoff;
  const unsigned short* Al = a_lo + aoff;
  const unsigned short* Bh = b_hi + boff;
  const unsigned short* Bl = b_lo + boff;

  const int arow[2] = { n_blk + wn +  0 + (lane & 15),
                        n_blk + wn + 16 + (lane & 15) };
  const int bcol[2] = { m_blk + wm +  0 + (lane & 15),
                        m_blk + wm + 16 + (lane & 15) };

  v8f acc[2][2] = {};
  for (int k0 = 0; k0 < D; k0 += 32) {
    v16bf fah[2], fal[2], fbh[2], fbl[2];
#pragma unroll
    for (int i = 0; i < 2; ++i) {
      fah[i] = load_frag_a(Ah, arow[i], k0, lane);
      fal[i] = load_frag_a(Al, arow[i], k0, lane);
      fbh[i] = load_frag_b(Bh, bcol[i], k0, lane);
      fbl[i] = load_frag_b(Bl, bcol[i], k0, lane);
    }
#pragma unroll
    for (int i = 0; i < 2; ++i)
#pragma unroll
      for (int j = 0; j < 2; ++j) {
        acc[i][j] = __builtin_amdgcn_wmma_f32_16x16x32_bf16(
            false, fal[i], false, fbh[j], (short)0, acc[i][j], false, false);
        acc[i][j] = __builtin_amdgcn_wmma_f32_16x16x32_bf16(
            false, fah[i], false, fbl[j], (short)0, acc[i][j], false, false);
        acc[i][j] = __builtin_amdgcn_wmma_f32_16x16x32_bf16(
            false, fah[i], false, fbh[j], (short)0, acc[i][j], false, false);
      }
  }

  // D layout: VGPR r, lanes 0-15 -> (row r, col lane); lanes 16-31 -> (row r+8, col lane-16)
  const int drow = (lane >> 4) * 8;
  const int dcol = lane & 15;

  // ---- store sim tile ----
  float* simb = sim + (size_t)bz * N * M;
#pragma unroll
  for (int i = 0; i < 2; ++i) {
    const int nbase = n_blk + wn + i * 16 + drow;
#pragma unroll
    for (int j = 0; j < 2; ++j) {
      const int mbase = m_blk + wm + j * 16 + dcol;
      float* out = simb + (size_t)nbase * M + mbase;
#pragma unroll
      for (int r = 0; r < 8; ++r)
        out[(size_t)r * M] = acc[i][j][r];
    }
  }

  // ---- fused reductions: per-lane stats -> LDS -> per-block partials ----
  __shared__ float r_mx[64][64], r_sm[64][64];  __shared__ int r_ai[64][64];
  __shared__ float c_mx[128][4], c_sm[128][4];  __shared__ int c_ai[128][4];

  // row side: each lane holds 2 cols (j=0,1) of 16 block-rows
  const int cg0 = m_blk + wm + dcol;        // global col, j = 0
  const int cg1 = cg0 + 16;                 // global col, j = 1
#pragma unroll
  for (int i = 0; i < 2; ++i)
#pragma unroll
    for (int r = 0; r < 8; ++r) {
      const int rowb = wn + i * 16 + drow + r;       // row within block (0..63)
      float v0 = acc[i][0][r], v1 = acc[i][1][r];
      float mx; float sm; int ai;
      if (v0 >= v1) { mx = v0; ai = cg0; sm = 1.f + __expf(v1 - v0); }
      else          { mx = v1; ai = cg1; sm = 1.f + __expf(v0 - v1); }
      const int ci = mw * 16 + dcol;                 // contributor 0..63
      r_mx[rowb][ci] = mx; r_sm[rowb][ci] = sm; r_ai[rowb][ci] = ai;
    }

  // col side: each lane holds 16 block-rows of 2 cols
#pragma unroll
  for (int j = 0; j < 2; ++j) {
    const int colb = wm + j * 16 + dcol;             // col within block (0..127)
    float mx = -3.402823466e38f, sm = 0.f; int ai = 0;
#pragma unroll
    for (int i = 0; i < 2; ++i)
#pragma unroll
      for (int r = 0; r < 8; ++r) {
        float v = acc[i][j][r];
        int rowg = n_blk + wn + i * 16 + drow + r;   // global row
        if (v > mx) { sm *= __expf(mx - v); mx = v; ai = rowg; }
        sm += __expf(v - mx);
      }
    const int ci = nw * 2 + (lane >> 4);             // contributor 0..3
    c_mx[colb][ci] = mx; c_sm[colb][ci] = sm; c_ai[colb][ci] = ai;
  }
  __syncthreads();

  if (tid < 64) {  // one thread per block-row: merge 64 contributors
    float mx = r_mx[tid][0], sm = r_sm[tid][0]; int ai = r_ai[tid][0];
    for (int k = 1; k < 64; ++k)
      osm_merge(mx, sm, ai, r_mx[tid][k], r_sm[tid][k], r_ai[tid][k]);
    const size_t p = (size_t)(bz * N + n_blk + tid) * MP + blockIdx.x;
    rp_mx[p] = mx; rp_sm[p] = sm; rp_ai[p] = ai;
  }
  if (tid < 128) { // one thread per block-col: merge 4 contributors
    float mx = c_mx[tid][0], sm = c_sm[tid][0]; int ai = c_ai[tid][0];
#pragma unroll
    for (int k = 1; k < 4; ++k)
      osm_merge(mx, sm, ai, c_mx[tid][k], c_sm[tid][k], c_ai[tid][k]);
    const size_t p = (size_t)(bz * M + m_blk + tid) * NP + blockIdx.y;
    cp_mx[p] = mx; cp_sm[p] = sm; cp_ai[p] = ai;
  }
}

// ---------------- finalize row partials: 32 chunks -> lse + argmax ----------------
__global__ __launch_bounds__(32) void row_final_kernel(
    const float* __restrict__ rp_mx, const float* __restrict__ rp_sm,
    const int* __restrict__ rp_ai, float* __restrict__ lse, int* __restrict__ arg) {
  const int row = blockIdx.x;                  // b*N + n
  const int lane = threadIdx.x;
  const size_t p = (size_t)row * MP + lane;
  float mx = rp_mx[p], sm = rp_sm[p]; int ai = rp_ai[p];
#pragma unroll
  for (int mask = 16; mask > 0; mask >>= 1) shfl_merge(mx, sm, ai, mask);
  if (lane == 0) { lse[row] = mx + __logf(sm); arg[row] = ai; }
}

// ---------------- finalize col partials: 64 chunks -> lse + argmax ----------------
__global__ __launch_bounds__(32) void col_final_kernel(
    const float* __restrict__ cp_mx, const float* __restrict__ cp_sm,
    const int* __restrict__ cp_ai, float* __restrict__ lse, int* __restrict__ arg) {
  const int col = blockIdx.x;                  // b*M + m
  const int lane = threadIdx.x;
  const size_t p = (size_t)col * NP + lane;
  float mx = cp_mx[p], sm = cp_sm[p]; int ai = cp_ai[p];
  osm_merge(mx, sm, ai, cp_mx[p + 32], cp_sm[p + 32], cp_ai[p + 32]);
#pragma unroll
  for (int mask = 16; mask > 0; mask >>= 1) shfl_merge(mx, sm, ai, mask);
  if (lane == 0) { lse[col] = mx + __logf(sm); arg[col] = ai; }
}

// ---------------- la = pad(2*sim - rowLSE - colLSE) ----------------
__global__ __launch_bounds__(256) void la_kernel(
    const float* __restrict__ sim, const float* __restrict__ rowlse,
    const float* __restrict__ collse, float* __restrict__ la) {
  const int bz = blockIdx.z;
  const int i  = blockIdx.y;                        // 0..N
  const int j  = blockIdx.x * 256 + threadIdx.x;    // 0..M
  if (j > M) return;
  float v = 0.f;
  if (i < N && j < M)
    v = 2.f * sim[(size_t)bz * N * M + (size_t)i * M + j]
        - rowlse[bz * N + i] - collse[bz * M + j];
  la[(size_t)bz * (N + 1) * (M + 1) + (size_t)i * (M + 1) + j] = v;
}

// ---------------- mutual check + scores + keypoint gathers ----------------
__global__ __launch_bounds__(256) void post_kernel(
    const int* __restrict__ rowarg, const int* __restrict__ colarg,
    const float* __restrict__ kpts0, const float* __restrict__ kpts1,
    int* __restrict__ m0, int* __restrict__ m1,
    float* __restrict__ ms0, float* __restrict__ ms1,
    float* __restrict__ mk0, float* __restrict__ mk1) {
  const int idx = blockIdx.x * 256 + threadIdx.x;   // b*N + n  (N == M)
  if (idx >= B * N) return;
  const int bz = idx / N, n = idx % N;
  // image0 -> image1
  const int r = rowarg[idx];                        // argmax is always valid
  const bool mut0 = (colarg[bz * M + r] == n);
  m0[idx]  = mut0 ? r : -1;
  ms0[idx] = mut0 ? 1.f : 0.f;
#pragma unroll
  for (int k = 0; k < 3; ++k) {
    mk0[(size_t)idx * 3 + k] = mut0 ? kpts0[(size_t)idx * 3 + k] : 0.f;
    mk1[(size_t)idx * 3 + k] = mut0 ? kpts1[((size_t)bz * M + r) * 3 + k] : 0.f;
  }
  // image1 -> image0
  const int c = colarg[idx];
  const bool mut1 = (rowarg[bz * N + c] == n);
  m1[idx]  = mut1 ? c : -1;
  ms1[idx] = mut1 ? 1.f : 0.f;
}

extern "C" void kernel_launch(void* const* d_in, const int* in_sizes, int n_in,
                              void* d_out, int out_size, void* d_ws, size_t ws_size,
                              hipStream_t stream) {
  (void)in_sizes; (void)n_in; (void)out_size; (void)ws_size;
  const float* desc0 = (const float*)d_in[0];
  const float* desc1 = (const float*)d_in[1];
  const float* kpts0 = (const float*)d_in[2];
  const float* kpts1 = (const float*)d_in[3];

  // ---- output partition (return order; all 4-byte elements) ----
  float* out = (float*)d_out;
  int*   m0  = (int*)out;                          out += (size_t)B * N;
  int*   m1  = (int*)out;                          out += (size_t)B * M;
  float* ms0 = out;                                out += (size_t)B * N;
  float* ms1 = out;                                out += (size_t)B * M;
  float* mk0 = out;                                out += (size_t)B * N * 3;
  float* mk1 = out;                                out += (size_t)B * N * 3;
  float* sim = out;                                out += (size_t)B * N * M;
  float* la  = out;

  // ---- workspace partition (~53 MB) ----
  unsigned short* d0h = (unsigned short*)d_ws;
  unsigned short* d0l = d0h + (size_t)B * N * D;
  unsigned short* d1h = d0l + (size_t)B * N * D;
  unsigned short* d1l = d1h + (size_t)B * M * D;
  float* rowlse = (float*)(d1l + (size_t)B * M * D);
  int*   rowarg = (int*)(rowlse + (size_t)B * N);
  float* collse = (float*)(rowarg + (size_t)B * N);
  int*   colarg = (int*)(collse + (size_t)B * M);
  float* rp_mx  = (float*)(colarg + (size_t)B * M);
  float* rp_sm  = rp_mx + (size_t)B * N * MP;
  int*   rp_ai  = (int*)(rp_sm + (size_t)B * N * MP);
  float* cp_mx  = (float*)(rp_ai + (size_t)B * N * MP);
  float* cp_sm  = cp_mx + (size_t)B * M * NP;
  int*   cp_ai  = (int*)(cp_sm + (size_t)B * M * NP);

  const int nd0 = B * N * D, nd1 = B * M * D;
  cvt_split_kernel<<<(nd0 + 255) / 256, 256, 0, stream>>>(desc0, d0h, d0l, nd0);
  cvt_split_kernel<<<(nd1 + 255) / 256, 256, 0, stream>>>(desc1, d1h, d1l, nd1);

  gemm_sim_kernel<<<dim3(M / 128, N / 64, B), 256, 0, stream>>>(
      d0h, d0l, d1h, d1l, sim, rp_mx, rp_sm, rp_ai, cp_mx, cp_sm, cp_ai);

  row_final_kernel<<<B * N, 32, 0, stream>>>(rp_mx, rp_sm, rp_ai, rowlse, rowarg);
  col_final_kernel<<<B * M, 32, 0, stream>>>(cp_mx, cp_sm, cp_ai, collse, colarg);

  la_kernel<<<dim3((M + 1 + 255) / 256, N + 1, B), 256, 0, stream>>>(sim, rowlse, collse, la);

  post_kernel<<<(B * N + 255) / 256, 256, 0, stream>>>(rowarg, colarg, kpts0, kpts1,
                                                       m0, m1, ms0, ms1, mk0, mk1);
}
